// LSSM_32100585570660
// MI455X (gfx1250) — compile-verified, matrix operand order checked by reading, not run on hardware
//
#include <hip/hip_runtime.h>

#define BATCH 16
#define SEQ   8192
#define DD    128
#define CH_L  64
#define NCH   (SEQ / CH_L)   // 128

typedef __attribute__((ext_vector_type(16))) __bf16 bf16x16;
typedef __attribute__((ext_vector_type(8)))  __bf16 bf16x8;
typedef __attribute__((ext_vector_type(8)))  float  f32x8;

static __device__ __forceinline__ f32x8 fzero() { f32x8 z = {}; return z; }

// Load a 16x32 bf16 WMMA fragment from a row-major [rows][128] bf16 array.
// Per-lane layout (ISA 7.12.2, 16-bit A/B matrix): lane = row (A: M, B: N),
// elements 0..7  -> k = k0 + 8*half + 0..7   (contiguous, 16 bytes)
// elements 8..15 -> k = k0 + 16 + 8*half + 0..7 (contiguous, 16 bytes)
static __device__ __forceinline__ bf16x16 load_frag(const __bf16* base, int row,
                                                    int k0, int half) {
  const __bf16* p = base + row * DD + k0 + half * 8;
  bf16x8 lo = *(const bf16x8*)p;
  bf16x8 hi = *(const bf16x8*)(p + 16);
  return __builtin_shufflevector(lo, hi, 0, 1, 2, 3, 4, 5, 6, 7,
                                 8, 9, 10, 11, 12, 13, 14, 15);
}

// acc += Amat[16x128] (rows = M) x Wmat[128 rows n][128 k] interpreted as Wᵀ.
// Per-lane: m = row for A-frag (lane&15), n = weight row for B-frag.
static __device__ __forceinline__ f32x8 mma_rowcol(const __bf16* Amat, int m,
                                                   const __bf16* Wmat, int n,
                                                   f32x8 acc, int half) {
#pragma unroll
  for (int kc = 0; kc < 4; ++kc) {
    bf16x16 a = load_frag(Amat, m, kc * 32, half);
    bf16x16 b = load_frag(Wmat, n, kc * 32, half);
    acc = __builtin_amdgcn_wmma_f32_16x16x32_bf16(false, a, false, b,
                                                  (short)0, acc, false, false);
  }
  return acc;
}

static __device__ __forceinline__ void copy32k(__bf16* dst, const __bf16* src,
                                               int tid) {
  const uint4* s = (const uint4*)src;
  uint4* d = (uint4*)dst;
#pragma unroll 4
  for (int i = tid; i < 2048; i += 256) d[i] = s[i];
}

// ---------------- Phase 1: fp32 weights -> bf16 (row-major [n][k]) ----------
__global__ void k_convert(const float* A, const float* B, const float* C,
                          const float* D, __bf16* Wa, __bf16* Wb, __bf16* Wc,
                          __bf16* Wd) {
  int idx = blockIdx.x * 256 + threadIdx.x;  // 0..65535
  int m = idx >> 14;
  int e = idx & 16383;
  const float* s = (m == 0) ? A : (m == 1) ? B : (m == 2) ? C : D;
  __bf16* d = (m == 0) ? Wa : (m == 1) ? Wb : (m == 2) ? Wc : Wd;
  d[e] = (__bf16)s[e];
}

// ---------------- Phase 2: P = A^64 (bf16, row-major) by squaring -----------
__global__ void k_power(const float* A, __bf16* Pst) {
  extern __shared__ char smem[];
  __bf16* G  = (__bf16*)smem;   // 32 KB
  __bf16* GT = G + 16384;       // 32 KB (transpose, for B-frag loads)
  __bf16* H  = GT + 16384;      // 32 KB
  __bf16* HT = H + 16384;       // 32 KB
  int tid = threadIdx.x, lane = tid & 31, wave = tid >> 5;
  int half = lane >> 4, lm = lane & 15;

  for (int i = tid; i < 16384; i += 256) {
    int r = i >> 7, c = i & 127;
    __bf16 v = (__bf16)A[i];
    G[r * 128 + c] = v;
    GT[c * 128 + r] = v;
  }
  __syncthreads();

  __bf16 *g = G, *gt = GT, *h = H, *ht = HT;
  for (int it = 0; it < 6; ++it) {  // 2^6 = 64 = CH_L
    int n = wave * 16 + lm;
    for (int mt = 0; mt < 8; ++mt) {
      int m = mt * 16 + lm;
      f32x8 acc = fzero();
      acc = mma_rowcol(g, m, gt, n, acc, half);
#pragma unroll
      for (int r = 0; r < 8; ++r) {
        int gm = mt * 16 + r + 8 * half;
        __bf16 v = (__bf16)acc[r];
        h[gm * 128 + n] = v;
        ht[n * 128 + gm] = v;
      }
    }
    __syncthreads();
    __bf16* t;
    t = g; g = h; h = t;
    t = gt; gt = ht; ht = t;
  }
  for (int i = tid; i < 8192; i += 256)
    ((unsigned int*)Pst)[i] = ((const unsigned int*)g)[i];
}

// ---------------- Phase 3: per-chunk carries r_c (zero entry state) ---------
__global__ void k_carry(const float* __restrict__ u, const __bf16* gWa,
                        const __bf16* gWb, float* __restrict__ R) {
  extern __shared__ char smem[];
  __bf16* Wa = (__bf16*)smem;     // 32 KB
  __bf16* Wb = Wa + 16384;        // 32 KB
  __bf16* X  = Wb + 16384;        // 2 x 4 KB (ping-pong state, bf16)
  __bf16* U  = X + 2 * 2048;      // 2 x 4 KB (ping-pong input, bf16)
  int tid = threadIdx.x, lane = tid & 31, wave = tid >> 5;
  int half = lane >> 4, lm = lane & 15;
  int c = blockIdx.x;

  copy32k(Wa, gWa, tid);
  copy32k(Wb, gWb, tid);
  for (int i = tid; i < 4096; i += 256) X[i] = (__bf16)0.0f;

  int b = tid >> 4;
  int kq = (tid & 15) * 8;
  const float* ubase = u + (size_t)b * SEQ * DD + kq;
  {  // preload u_{cL} into U buffer 0
    const float4* p = (const float4*)(ubase + (size_t)(c * CH_L) * DD);
    float4 f0 = p[0], f1 = p[1];
    __bf16* Ud = U + b * DD + kq;
    Ud[0] = (__bf16)f0.x; Ud[1] = (__bf16)f0.y;
    Ud[2] = (__bf16)f0.z; Ud[3] = (__bf16)f0.w;
    Ud[4] = (__bf16)f1.x; Ud[5] = (__bf16)f1.y;
    Ud[6] = (__bf16)f1.z; Ud[7] = (__bf16)f1.w;
  }
  __syncthreads();

  int n = wave * 16 + lm;
  f32x8 acc = fzero();
  for (int j = 0; j < CH_L; ++j) {
    __bf16* Xcur = X + (j & 1) * 2048;
    __bf16* Xnxt = X + ((j + 1) & 1) * 2048;
    __bf16* Ucur = U + (j & 1) * 2048;
    __bf16* Unxt = U + ((j + 1) & 1) * 2048;

    float4 f0 = {}, f1 = {};
    if (j + 1 < CH_L) {
      const float4* p = (const float4*)(ubase + (size_t)(c * CH_L + j + 1) * DD);
      f0 = p[0];
      f1 = p[1];
    }
    if (j + 2 < CH_L)
      __builtin_prefetch(ubase + (size_t)(c * CH_L + j + 2) * DD, 0, 0);

    acc = fzero();
    acc = mma_rowcol(Xcur, lm, Wa, n, acc, half);  // x·Aᵀ
    acc = mma_rowcol(Ucur, lm, Wb, n, acc, half);  // + u·Bᵀ
#pragma unroll
    for (int r = 0; r < 8; ++r)
      Xnxt[(r + 8 * half) * DD + n] = (__bf16)acc[r];
    if (j + 1 < CH_L) {
      __bf16* Ud = Unxt + b * DD + kq;
      Ud[0] = (__bf16)f0.x; Ud[1] = (__bf16)f0.y;
      Ud[2] = (__bf16)f0.z; Ud[3] = (__bf16)f0.w;
      Ud[4] = (__bf16)f1.x; Ud[5] = (__bf16)f1.y;
      Ud[6] = (__bf16)f1.z; Ud[7] = (__bf16)f1.w;
    }
    __syncthreads();
  }
  // final acc = state at end of chunk = r_c
#pragma unroll
  for (int r = 0; r < 8; ++r)
    R[((size_t)c * BATCH + r + 8 * half) * DD + n] = acc[r];
}

// ---------------- Phase 4: sequential chunk-level scan ----------------------
__global__ void k_scan(const float* __restrict__ R, const __bf16* gP,
                       float* __restrict__ S) {
  extern __shared__ char smem[];
  __bf16* P = (__bf16*)smem;  // 32 KB  (Pst = A^L row-major -> B-frag layout)
  __bf16* X = P + 16384;      // 2 x 4 KB
  int tid = threadIdx.x, lane = tid & 31, wave = tid >> 5;
  int half = lane >> 4, lm = lane & 15;

  copy32k(P, gP, tid);
  for (int i = tid; i < 4096; i += 256) X[i] = (__bf16)0.0f;
  for (int i = tid; i < 2048; i += 256) S[i] = 0.0f;  // entry state 0 = 0
  __syncthreads();

  int n = wave * 16 + lm;
  for (int cc = 1; cc < NCH; ++cc) {
    __bf16* Xcur = X + ((cc - 1) & 1) * 2048;
    __bf16* Xnxt = X + (cc & 1) * 2048;
    f32x8 acc;
#pragma unroll
    for (int r = 0; r < 8; ++r)  // C operand = r_{c-1}: free add via WMMA
      acc[r] = R[((size_t)(cc - 1) * BATCH + r + 8 * half) * DD + n];
    acc = mma_rowcol(Xcur, lm, P, n, acc, half);  // entry_{c-1}·(Aᵀ)^L + r_{c-1}
#pragma unroll
    for (int r = 0; r < 8; ++r) {
      S[((size_t)cc * BATCH + r + 8 * half) * DD + n] = acc[r];
      Xnxt[(r + 8 * half) * DD + n] = (__bf16)acc[r];
    }
    __syncthreads();
  }
}

// ---------------- Phase 5: re-run chunks from true entry, emit y ------------
__global__ void k_out(const float* __restrict__ u, const float* __restrict__ S,
                      const __bf16* gWa, const __bf16* gWb, const __bf16* gWc,
                      const __bf16* gWd, float* __restrict__ y) {
  extern __shared__ char smem[];
  __bf16* Wa = (__bf16*)smem;
  __bf16* Wb = Wa + 16384;
  __bf16* Wc = Wb + 16384;
  __bf16* Wd = Wc + 16384;
  __bf16* X  = Wd + 16384;   // 2 x 4 KB
  __bf16* U  = X + 2 * 2048; // 2 x 4 KB
  int tid = threadIdx.x, lane = tid & 31, wave = tid >> 5;
  int half = lane >> 4, lm = lane & 15;
  int c = blockIdx.x;

  copy32k(Wa, gWa, tid);
  copy32k(Wb, gWb, tid);
  copy32k(Wc, gWc, tid);
  copy32k(Wd, gWd, tid);
  for (int i = tid; i < 2048; i += 256)  // entry state for this chunk
    X[i] = (__bf16)S[(size_t)c * 2048 + i];

  int b = tid >> 4;
  int kq = (tid & 15) * 8;
  const float* ubase = u + (size_t)b * SEQ * DD + kq;
  {
    const float4* p = (const float4*)(ubase + (size_t)(c * CH_L) * DD);
    float4 f0 = p[0], f1 = p[1];
    __bf16* Ud = U + b * DD + kq;
    Ud[0] = (__bf16)f0.x; Ud[1] = (__bf16)f0.y;
    Ud[2] = (__bf16)f0.z; Ud[3] = (__bf16)f0.w;
    Ud[4] = (__bf16)f1.x; Ud[5] = (__bf16)f1.y;
    Ud[6] = (__bf16)f1.z; Ud[7] = (__bf16)f1.w;
  }
  __syncthreads();

  int n = wave * 16 + lm;
  for (int j = 0; j < CH_L; ++j) {
    __bf16* Xcur = X + (j & 1) * 2048;
    __bf16* Xnxt = X + ((j + 1) & 1) * 2048;
    __bf16* Ucur = U + (j & 1) * 2048;
    __bf16* Unxt = U + ((j + 1) & 1) * 2048;

    float4 f0 = {}, f1 = {};
    if (j + 1 < CH_L) {
      const float4* p = (const float4*)(ubase + (size_t)(c * CH_L + j + 1) * DD);
      f0 = p[0];
      f1 = p[1];
    }
    if (j + 2 < CH_L)
      __builtin_prefetch(ubase + (size_t)(c * CH_L + j + 2) * DD, 0, 0);

    f32x8 sacc = fzero();
    sacc = mma_rowcol(Xcur, lm, Wa, n, sacc, half);  // x·Aᵀ
    sacc = mma_rowcol(Ucur, lm, Wb, n, sacc, half);  // + u·Bᵀ
    f32x8 yacc = fzero();
    yacc = mma_rowcol(Ucur, lm, Wd, n, yacc, half);  // u·Dᵀ

#pragma unroll
    for (int r = 0; r < 8; ++r)
      Xnxt[(r + 8 * half) * DD + n] = (__bf16)sacc[r];
    if (j + 1 < CH_L) {
      __bf16* Ud = Unxt + b * DD + kq;
      Ud[0] = (__bf16)f0.x; Ud[1] = (__bf16)f0.y;
      Ud[2] = (__bf16)f0.z; Ud[3] = (__bf16)f0.w;
      Ud[4] = (__bf16)f1.x; Ud[5] = (__bf16)f1.y;
      Ud[6] = (__bf16)f1.z; Ud[7] = (__bf16)f1.w;
    }
    __syncthreads();  // Xnxt + Unxt visible; single barrier per step

    yacc = mma_rowcol(Xnxt, lm, Wc, n, yacc, half);  // + x_t·Cᵀ
    size_t t = (size_t)c * CH_L + j;
#pragma unroll
    for (int r = 0; r < 8; ++r)
      y[((size_t)(r + 8 * half) * SEQ + t) * DD + n] = yacc[r];
  }
}

extern "C" void kernel_launch(void* const* d_in, const int* in_sizes, int n_in,
                              void* d_out, int out_size, void* d_ws,
                              size_t ws_size, hipStream_t stream) {
  const float* u = (const float*)d_in[0];
  const float* A = (const float*)d_in[1];
  const float* B = (const float*)d_in[2];
  const float* C = (const float*)d_in[3];
  const float* D = (const float*)d_in[4];
  float* y = (float*)d_out;

  char* ws = (char*)d_ws;
  __bf16* Wa = (__bf16*)ws;
  __bf16* Wb = Wa + 16384;
  __bf16* Wc = Wb + 16384;
  __bf16* Wd = Wc + 16384;
  __bf16* P  = Wd + 16384;
  float* R = (float*)(ws + 5 * 32768);       // [NCH][16][128] carries
  float* S = R + (size_t)NCH * BATCH * DD;   // [NCH][16][128] entry states

  k_convert<<<256, 256, 0, stream>>>(A, B, C, D, Wa, Wb, Wc, Wd);
  k_power<<<1, 256, 4 * 32768, stream>>>(A, P);
  k_carry<<<NCH, 256, 2 * 32768 + 4 * 4096, stream>>>(u, Wa, Wb, R);
  k_scan<<<1, 256, 32768 + 2 * 4096, stream>>>(R, P, S);
  k_out<<<NCH, 256, 4 * 32768 + 4 * 4096, stream>>>(u, S, Wa, Wb, Wc, Wd, y);
}
